// DummyModel_73916387164337
// MI455X (gfx1250) — matrix operand order repaired and checked by reading, not compiled, not run on hardware
//
#include <hip/hip_runtime.h>
#include <stdint.h>
#include <stddef.h>

typedef _Float16 v16h  __attribute__((ext_vector_type(16)));
typedef __fp16   fp16x2 __attribute__((ext_vector_type(2)));
typedef float    v8f   __attribute__((ext_vector_type(8)));

#define N_IN   4096
#define N_OUT  8192
#define TPRED  4
#define WAVES  4
// -log2(e) / (2 * LENGTH_SCALE^2), LENGTH_SCALE = 0.1
#define KEXP   (-72.13475204444817f)
#define EPSV   (1e-8f)

union AU { v16h v; fp16x2 p[8]; };
union BU { v16h v; unsigned u[8]; };

__launch_bounds__(128)
__global__ void rbf_wmma_kernel(const float* __restrict__ input_data,
                                const float* __restrict__ input_coords,
                                const float* __restrict__ target_coords,
                                float* __restrict__ out,
                                int B, int T_in, int ns)
{
    __shared__ float2 s_coords[N_IN];                 // 32 KB
    __shared__ __align__(16) _Float16 s_vals[N_IN];   // 8 KB
    __shared__ float s_num[WAVES][16];
    __shared__ float s_den[WAVES][16];

    const int tid  = threadIdx.x;        // 0..127
    const int wv   = tid >> 5;           // wave id in block
    const int lane = tid & 31;           // wave32 lane
    const int b    = blockIdx.y;

    // ---- cooperative preload of this batch's inputs into LDS ----
    const float2* ic = (const float2*)(input_coords + (size_t)b * N_IN * 2);
    // input_data[b, T_in-1, i, 0] with V_in == 3
    const float* vbase = input_data + (((size_t)b * T_in + (T_in - 1)) * N_IN) * 3;
    for (int i = tid; i < N_IN; i += 128) {
        s_coords[i] = ic[i];
        s_vals[i]   = (_Float16)vbase[(size_t)i * 3];
    }
    __syncthreads();

    // ---- per-wave 16-output tile ----
    const int tile_base = (blockIdx.x * WAVES + wv) * 16;
    const int m = lane & 15;             // A-row (lanes 0-15 and 16-31 mirror M)
    const float2 tc =
        ((const float2*)(target_coords + (size_t)b * N_OUT * 2))[tile_base + m];
    const float tx = tc.x, ty = tc.y;

    // A-matrix K base, contiguous order (lanes 0-15: K=0-15, lanes 16-31: K=16-31)
    const int kb_a = (lane & 16) ? 16 : 0;
    // B-matrix K base, contiguous order as in A (lanes 0-15: K=0-15, lanes 16-31: K=16-31)
    const int kb_b = (lane & 16) ? 16 : 0;
    // B column index = lane & 15; column 0 = values, column 1 = ones, rest 0
    const unsigned bfill = (m == 1) ? 0x3C003C00u : 0u;  // packed f16 {1.0,1.0} or 0
    BU bu0, bu1;
#pragma unroll
    for (int j = 0; j < 8; ++j) { bu0.u[j] = bfill; bu1.u[j] = bfill; }

    v8f acc0 = {0.f, 0.f, 0.f, 0.f, 0.f, 0.f, 0.f, 0.f};
    v8f acc1 = {0.f, 0.f, 0.f, 0.f, 0.f, 0.f, 0.f, 0.f};

    // 2x unrolled K loop: two independent A-fragments + accumulators so the
    // second half's exp/cvt VALU chain fills the first WMMA's WAR hazard
    // window (no v_nop padding) and the matrix pipe sees two dep chains.
    for (int c = 0; c < N_IN; c += 64) {
        AU au0, au1;
#pragma unroll
        for (int h = 0; h < 2; ++h) {
#pragma unroll
            for (int jj = 0; jj < 4; ++jj) {
                const int k0 = c + kb_a + h * 8 + 2 * jj;
                {
                    const float2 p0 = s_coords[k0];
                    const float2 p1 = s_coords[k0 + 1];
                    const float dx0 = tx - p0.x, dy0 = ty - p0.y;
                    const float dx1 = tx - p1.x, dy1 = ty - p1.y;
                    const float e0 =
                        __builtin_amdgcn_exp2f(fmaf(dx0, dx0, dy0 * dy0) * KEXP);
                    const float e1 =
                        __builtin_amdgcn_exp2f(fmaf(dx1, dx1, dy1 * dy1) * KEXP);
                    au0.p[h * 4 + jj] = __builtin_amdgcn_cvt_pkrtz(e0, e1);
                }
                {
                    const float2 p0 = s_coords[k0 + 32];
                    const float2 p1 = s_coords[k0 + 33];
                    const float dx0 = tx - p0.x, dy0 = ty - p0.y;
                    const float dx1 = tx - p1.x, dy1 = ty - p1.y;
                    const float e0 =
                        __builtin_amdgcn_exp2f(fmaf(dx0, dx0, dy0 * dy0) * KEXP);
                    const float e1 =
                        __builtin_amdgcn_exp2f(fmaf(dx1, dx1, dy1 * dy1) * KEXP);
                    au1.p[h * 4 + jj] = __builtin_amdgcn_cvt_pkrtz(e0, e1);
                }
            }
        }
        // ---- refresh B column 0 (values) for both halves ----
        if (m == 0) {
            const unsigned* vp0 = (const unsigned*)(s_vals + c + kb_b);       // 4B aligned
            const unsigned* vp1 = (const unsigned*)(s_vals + c + 32 + kb_b);
#pragma unroll
            for (int j = 0; j < 8; ++j) { bu0.u[j] = vp0[j]; bu1.u[j] = vp1[j]; }
        }
        // D[:,0] += W * v ; D[:,1] += W * 1   (f32 accumulate)
        acc0 = __builtin_amdgcn_wmma_f32_16x16x32_f16(
            false, au0.v, false, bu0.v, (short)0, acc0, false, false);
        acc1 = __builtin_amdgcn_wmma_f32_16x16x32_f16(
            false, au1.v, false, bu1.v, (short)0, acc1, false, false);
    }

    const v8f acc = acc0 + acc1;

    // ---- gather numerator (N=0 -> lanes 0,16) and denominator (N=1 -> lanes 1,17) ----
    if (lane == 0 || lane == 16) {
        const int off = (lane >> 4) * 8;   // lanes 16-31 hold M=8..15
#pragma unroll
        for (int r = 0; r < 8; ++r) s_num[wv][off + r] = acc[r];
    }
    if (lane == 1 || lane == 17) {
        const int off = (lane >> 4) * 8;
#pragma unroll
        for (int r = 0; r < 8; ++r) s_den[wv][off + r] = acc[r];
    }
    __syncthreads();

    // ---- normalize and broadcast-store: out[b, s, o, t] = val for all s,t ----
    if (lane < 16) {
        const float val = s_num[wv][lane] / (s_den[wv][lane] + EPSV);
        const float4 v4 = make_float4(val, val, val, val);
        float4* o4 = (float4*)out;                   // TPRED==4 floats per (b,s,o)
        const size_t o = (size_t)tile_base + lane;
        for (int s = 0; s < ns; ++s)
            o4[(size_t)(b * ns + s) * N_OUT + o] = v4;
    }
}

extern "C" void kernel_launch(void* const* d_in, const int* in_sizes, int n_in,
                              void* d_out, int out_size, void* d_ws, size_t ws_size,
                              hipStream_t stream) {
    const float* input_data    = (const float*)d_in[0];
    const float* input_coords  = (const float*)d_in[1];
    const float* target_coords = (const float*)d_in[2];
    (void)d_ws; (void)ws_size; (void)n_in;

    // Derive shapes defensively from the harness metadata.
    int B = in_sizes[2] / (N_OUT * 2);          // target_coords = (B, N_out, 2)
    if (B <= 0) B = 8;
    int T_in = in_sizes[0] / (B * N_IN * 3);    // input_data = (B, T_in, N_in, 3)
    if (T_in <= 0) T_in = 4;
    int ns = out_size / (B * N_OUT * TPRED);    // output = (B, ns, N_out, TPRED)
    if (ns <= 0) ns = 1;

    dim3 grid(N_OUT / (16 * WAVES), B);         // (128, B); 4 waves/block, 16 outputs/wave
    rbf_wmma_kernel<<<grid, 128, 0, stream>>>(input_data, input_coords, target_coords,
                                              (float*)d_out, B, T_in, ns);
}
